// Polarizable_13872744366325
// MI455X (gfx1250) — matrix operand order repaired and checked by reading, not compiled, not run on hardware
//
#include <hip/hip_runtime.h>
#include <hip/hip_bf16.h>
#include <math.h>

typedef __attribute__((ext_vector_type(16))) _Float16 v16h;
typedef __attribute__((ext_vector_type(8)))  _Float16 v8h;
typedef __attribute__((ext_vector_type(8)))  float    v8f;

#define NN     5000
#define EE     160000
#define CC     64
#define NELEM  10
#define RMAXF  5.0f
#define AVGN   32.0f

__device__ __forceinline__ float silu_f(float x) {
  // x * sigmoid(x); fast v_rcp_f32 instead of IEEE division
  return x * __builtin_amdgcn_rcpf(1.0f + __expf(-x));
}

__device__ __forceinline__ v8f wmma_f16(v16h a, v16h b, v8f c) {
  // D(16x16 f32) = A(16x32 f16) * B(32x16 f16) + C
  return __builtin_amdgcn_wmma_f32_16x16x32_f16(false, a, false, b, (short)0, c, false, false);
}

// A-fragment: 16x32 f16 tile from row-major [16 x ld] f16 buffer, K window [koff, koff+32).
// ISA 7.12.2 layout = two contiguous 8-half runs per lane -> two 16B vector loads.
__device__ __forceinline__ v16h make_a(const _Float16* A, int ld, int lane, int koff) {
  const int row = lane & 15;
  const int kb  = (lane & 16) ? 8 : 0;
  const _Float16* p = A + row * ld + koff + kb;
  v8h lo = *(const v8h*)p;          // K = koff+kb   .. +7
  v8h hi = *(const v8h*)(p + 16);   // K = koff+16+kb .. +7
  return __builtin_shufflevector(lo, hi, 0, 1, 2, 3, 4, 5, 6, 7,
                                 8, 9, 10, 11, 12, 13, 14, 15);
}

// B-fragment: 32x16 f16. Lane%16 = N column, vector elements = K in the same order as make_a.
__device__ __forceinline__ v16h make_b(const _Float16* B, int ld, int lane, int noff) {
  const int col = lane & 15;
  const int kb  = (lane & 16) ? 8 : 0;
  v16h b;
#pragma unroll
  for (int s = 0; s < 8; ++s) {
    b[s]     = B[(size_t)(kb + s) * ld + noff + col];
    b[8 + s] = B[(size_t)(16 + kb + s) * ld + noff + col];
  }
  return b;
}

// ---------------------------------------------------------------------------
// Weight conversion f32 -> f16 scratch
__global__ void k_prep(const float* __restrict__ Wr1, const float* __restrict__ Wr2,
                       const float* __restrict__ Wr3, const float* __restrict__ Wprod,
                       _Float16* __restrict__ Wr1h, _Float16* __restrict__ Wr2h,
                       _Float16* __restrict__ Wr3h, _Float16* __restrict__ Wprodh) {
  int gid = blockIdx.x * blockDim.x + threadIdx.x;
  if (gid < 2 * 8 * 64)   Wr1h[gid]   = (_Float16)Wr1[gid];
  if (gid < 2 * 64 * 64)  Wr2h[gid]   = (_Float16)Wr2[gid];
  if (gid < 2 * 64 * 192) Wr3h[gid]   = (_Float16)Wr3[gid];
  if (gid < 2 * 192 * 64) Wprodh[gid] = (_Float16)Wprod[gid];
}

// h = node_attrs @ W_embed ; e_node = node_attrs @ atomic_energies ; q=0 ; energy=0
__global__ void k_embed(const float* __restrict__ node_attrs, const float* __restrict__ W_embed,
                        const float* __restrict__ atomic_energies, float* __restrict__ h,
                        float* __restrict__ energy, float* __restrict__ e_node,
                        float* __restrict__ q) {
  int gid = blockIdx.x * blockDim.x + threadIdx.x;
  if (gid < 4) energy[gid] = 0.0f;
  if (gid >= NN * CC) return;
  int n = gid >> 6, c = gid & 63;
  const float* ap = node_attrs + n * NELEM;
  float acc = 0.0f;
#pragma unroll
  for (int k = 0; k < NELEM; ++k) acc += ap[k] * W_embed[k * CC + c];
  h[gid] = acc;
  if (c == 0) {
    float ea = 0.0f;
#pragma unroll
    for (int k = 0; k < NELEM; ++k) ea += ap[k] * atomic_energies[k];
    e_node[n] = ea;
    q[n] = 0.0f;
  }
}

__global__ void k_zero4(float4* __restrict__ p, int n4) {
  int i = blockIdx.x * blockDim.x + threadIdx.x;
  if (i < n4) p[i] = float4{0.f, 0.f, 0.f, 0.f};
}

// ---------------------------------------------------------------------------
// Fused edge kernel: geometry -> radial/SH -> MLP(8->64->64->192 via WMMA) -> msg scatter
__global__ __launch_bounds__(256) void k_edge(const float* __restrict__ positions,
                                              const float* __restrict__ shifts,
                                              const int* __restrict__ edge_index,
                                              const float* __restrict__ h,
                                              const _Float16* __restrict__ Wr1h,   // 8x64
                                              const _Float16* __restrict__ Wr2h,   // 64x64
                                              const _Float16* __restrict__ Wr3h,   // 64x192
                                              float* __restrict__ m) {
  __shared__ __align__(32) _Float16 sWr1[32 * 64];   // zero-padded K: rows 8..31 = 0
  __shared__ __align__(32) _Float16 sWr2[64 * 64];
  __shared__ __align__(32) _Float16 sX[8][16 * 64];  // activations (GEMM1 uses cols 0..31)
  __shared__ __align__(32) _Float16 sH[8][16 * 64];
  __shared__ float sY[8][16 * 9];
  __shared__ int   sSend[8][16];
  __shared__ int   sRecv[8][16];

  const int tid  = threadIdx.x;
  const int wid  = tid >> 5;
  const int lane = tid & 31;
  const int eBase = blockIdx.x * 128 + wid * 16;

  // warm L2/L0 with the streamed GEMM3 weights (global_prefetch_b8)
  if (tid < (64 * 192 * 2) / 128)
    __builtin_prefetch((const char*)Wr3h + tid * 128, 0, 1);

  for (int idx = tid; idx < 32 * 64; idx += 256)
    sWr1[idx] = (idx < 8 * 64) ? Wr1h[idx] : (_Float16)0.0f;
  for (int idx = tid; idx < 64 * 64; idx += 256) sWr2[idx] = Wr2h[idx];

  // --- geometry: lanes 0..15 each own one edge of the 16-edge tile ---
  if (lane < 16) {
    int e = eBase + lane;
    float Y[9], ef[8];
#pragma unroll
    for (int s = 0; s < 9; ++s) Y[s] = 0.0f;
#pragma unroll
    for (int k = 0; k < 8; ++k) ef[k] = 0.0f;
    int snd = -1, rcv = -1;
    if (e < EE) {
      snd = edge_index[e];
      rcv = edge_index[EE + e];
      float vx = positions[rcv * 3 + 0] - positions[snd * 3 + 0] + shifts[e * 3 + 0];
      float vy = positions[rcv * 3 + 1] - positions[snd * 3 + 1] + shifts[e * 3 + 1];
      float vz = positions[rcv * 3 + 2] - positions[snd * 3 + 2] + shifts[e * 3 + 2];
      float rr = fmaxf(sqrtf(vx * vx + vy * vy + vz * vz), 1e-9f);
      float ir = 1.0f / rr;
      float x = vx * ir, y = vy * ir, z = vz * ir;
      const float s3 = 1.7320508075688772f, s5 = 2.2360679774997896f, s15 = 3.872983346207417f;
      Y[0] = 1.0f;        Y[1] = s3 * x;      Y[2] = s3 * y;      Y[3] = s3 * z;
      Y[4] = s15 * x * y; Y[5] = s15 * y * z; Y[6] = 0.5f * s5 * (3.0f * z * z - 1.0f);
      Y[7] = s15 * x * z; Y[8] = 0.5f * s15 * (x * x - y * y);
      float u = rr * (1.0f / RMAXF);
      float u2 = u * u, u4 = u2 * u2, u5 = u4 * u, u6 = u4 * u2, u7 = u6 * u;
      float fcut = (u < 1.0f) ? (1.0f - 21.0f * u5 + 35.0f * u6 - 15.0f * u7) : 0.0f;
      float coef = sqrtf(2.0f / RMAXF) * ir * fcut;
#pragma unroll
      for (int k = 0; k < 8; ++k)
        ef[k] = coef * __sinf(3.14159265358979f * (float)(k + 1) * u);
    }
#pragma unroll
    for (int s = 0; s < 9; ++s) sY[wid][lane * 9 + s] = Y[s];
    // ef into sX cols 0..7, zeros in cols 8..31 (GEMM1 A window is K=0..31)
    v8h efv, zv;
#pragma unroll
    for (int k = 0; k < 8; ++k) { efv[k] = (_Float16)ef[k]; zv[k] = (_Float16)0.0f; }
    _Float16* xp = &sX[wid][lane * 64];
    *(v8h*)(xp + 0)  = efv;
    *(v8h*)(xp + 8)  = zv;
    *(v8h*)(xp + 16) = zv;
    *(v8h*)(xp + 24) = zv;
    sSend[wid][lane] = snd;
    sRecv[wid][lane] = rcv;
  }
  __syncthreads();

  // --- gather h[send] rows for the tile ---
  for (int t = 0; t < 16; ++t) {
    int snd = sSend[wid][t];
    const float* hp = h + (size_t)(snd < 0 ? 0 : snd) * CC;
    for (int c = lane; c < CC; c += 32)
      sH[wid][t * CC + c] = (snd >= 0) ? (_Float16)hp[c] : (_Float16)0.0f;
  }
  __syncthreads();

  const int row0 = (lane & 16) ? 8 : 0;
  const int col  = lane & 15;

  // --- GEMM1: ef(16x8, zero-padded to K=32) @ Wr1(32x64 padded) -> silu -> sX ---
  {
    v16h a = make_a(sX[wid], 64, lane, 0);
    v8f acc[4];
#pragma unroll
    for (int nt = 0; nt < 4; ++nt) {
      v8f z = {};
      acc[nt] = wmma_f16(a, make_b(sWr1, 64, lane, nt * 16), z);
    }
    __syncthreads();   // fragment reads done before overwrite
#pragma unroll
    for (int nt = 0; nt < 4; ++nt)
#pragma unroll
      for (int v = 0; v < 8; ++v)
        sX[wid][(v + row0) * 64 + col + nt * 16] = (_Float16)silu_f(acc[nt][v]);
  }
  __syncthreads();

  // --- GEMM2: x(16x64) @ Wr2(64x64) -> silu -> sX ---
  {
    v16h a0 = make_a(sX[wid], 64, lane, 0);
    v16h a1 = make_a(sX[wid], 64, lane, 32);
    v8f acc[4];
#pragma unroll
    for (int nt = 0; nt < 4; ++nt) {
      v8f z = {};
      z = wmma_f16(a0, make_b(sWr2, 64, lane, nt * 16), z);
      z = wmma_f16(a1, make_b(sWr2 + 32 * 64, 64, lane, nt * 16), z);
      acc[nt] = z;
    }
    __syncthreads();   // all reads of sX done before overwrite
#pragma unroll
    for (int nt = 0; nt < 4; ++nt)
#pragma unroll
      for (int v = 0; v < 8; ++v)
        sX[wid][(v + row0) * 64 + col + nt * 16] = (_Float16)silu_f(acc[nt][v]);
  }
  __syncthreads();

  // --- GEMM3: x(16x64) @ Wr3(64x192), then msg = h[send]*Y*w scattered to m[recv] ---
  {
    v16h a0 = make_a(sX[wid], 64, lane, 0);
    v16h a1 = make_a(sX[wid], 64, lane, 32);
    for (int nt = 0; nt < 12; ++nt) {
      v8f acc = {};
      acc = wmma_f16(a0, make_b(Wr3h, 192, lane, nt * 16), acc);
      acc = wmma_f16(a1, make_b(Wr3h + 32 * 192, 192, lane, nt * 16), acc);
      const int feat = col + nt * 16;           // 0..191
      const int cch  = feat / 3;                // channel
      const int l    = feat - cch * 3;          // 0,1,2 -> LIDX groups {0},{1,2,3},{4..8}
#pragma unroll
      for (int v = 0; v < 8; ++v) {
        int row = v + row0;
        int rcv = sRecv[wid][row];
        if (rcv < 0) continue;
        float hwv = (float)sH[wid][row * CC + cch] * acc[v];
        float* dst = m + (size_t)rcv * 576 + cch * 9;
        const float* Yp = &sY[wid][row * 9];
        if (l == 0) {
          atomicAdd(dst + 0, hwv * Yp[0]);
        } else if (l == 1) {
          atomicAdd(dst + 1, hwv * Yp[1]);
          atomicAdd(dst + 2, hwv * Yp[2]);
          atomicAdd(dst + 3, hwv * Yp[3]);
        } else {
          atomicAdd(dst + 4, hwv * Yp[4]);
          atomicAdd(dst + 5, hwv * Yp[5]);
          atomicAdd(dst + 6, hwv * Yp[6]);
          atomicAdd(dst + 7, hwv * Yp[7]);
          atomicAdd(dst + 8, hwv * Yp[8]);
        }
      }
    }
  }
}

// ---------------------------------------------------------------------------
// Node update: invariants -> inv(16x192) @ Wprod(192x64) via WMMA, residual h
__global__ __launch_bounds__(256) void k_node(const float* __restrict__ m,
                                              const _Float16* __restrict__ Wprodh, // 192x64
                                              float* __restrict__ h, int iter) {
  __shared__ __align__(32) _Float16 sInv[8][16 * 192];
  const int tid = threadIdx.x, wid = tid >> 5, lane = tid & 31;
  const int nb = blockIdx.x * 128 + wid * 16;

  for (int idx = lane; idx < 16 * 64; idx += 32) {
    int t = idx >> 6, c = idx & 63;
    int n = nb + t;
    float i0 = 0.f, i1 = 0.f, i2 = 0.f;
    if (n < NN) {
      const float* mp = m + (size_t)n * 576 + c * 9;
      float a0 = mp[0] * (1.f / AVGN);
      float a1 = mp[1] * (1.f / AVGN), a2 = mp[2] * (1.f / AVGN), a3 = mp[3] * (1.f / AVGN);
      float a4 = mp[4] * (1.f / AVGN), a5 = mp[5] * (1.f / AVGN), a6 = mp[6] * (1.f / AVGN);
      float a7 = mp[7] * (1.f / AVGN), a8 = mp[8] * (1.f / AVGN);
      i0 = a0 * a0;
      i1 = a1 * a1 + a2 * a2 + a3 * a3;
      i2 = a4 * a4 + a5 * a5 + a6 * a6 + a7 * a7 + a8 * a8;
    }
    sInv[wid][t * 192 + c * 3 + 0] = (_Float16)i0;
    sInv[wid][t * 192 + c * 3 + 1] = (_Float16)i1;
    sInv[wid][t * 192 + c * 3 + 2] = (_Float16)i2;
  }
  __syncthreads();

  v16h a[6];
#pragma unroll
  for (int kc = 0; kc < 6; ++kc) a[kc] = make_a(sInv[wid], 192, lane, kc * 32);

  const int row0 = (lane & 16) ? 8 : 0;
#pragma unroll
  for (int nt = 0; nt < 4; ++nt) {
    v8f acc = {};
#pragma unroll
    for (int kc = 0; kc < 6; ++kc)
      acc = wmma_f16(a[kc], make_b(Wprodh + (size_t)kc * 32 * 64, 64, lane, nt * 16), acc);
    const int ch = (lane & 15) + nt * 16;
#pragma unroll
    for (int v = 0; v < 8; ++v) {
      int n = nb + v + row0;
      if (n < NN) {
        float m0 = m[(size_t)n * 576 + ch * 9] * (1.f / AVGN);
        float hn = acc[v] + m0;                          // h_new = inv@Wprod + m[:,:,0]
        h[n * CC + ch] = (iter == 0) ? hn : (hn + h[n * CC + ch]);
      }
    }
  }
}

// Readouts: e_node += readout(h), q += h @ Wq[i]
__global__ void k_readout(const float* __restrict__ h, const float* __restrict__ w_read0,
                          const float* __restrict__ read_W1, const float* __restrict__ read_W2,
                          const float* __restrict__ Wq_i, float* __restrict__ e_node,
                          float* __restrict__ q, int iter) {
  int n = blockIdx.x * blockDim.x + threadIdx.x;
  if (n >= NN) return;
  const float* hp = h + (size_t)n * CC;
  float qa = 0.0f;
#pragma unroll 8
  for (int c = 0; c < CC; ++c) qa += hp[c] * Wq_i[c];
  q[n] += qa;
  float ea = 0.0f;
  if (iter == 0) {
#pragma unroll 8
    for (int c = 0; c < CC; ++c) ea += hp[c] * w_read0[c];
  } else {
    float t[16];
#pragma unroll
    for (int j = 0; j < 16; ++j) t[j] = 0.0f;
    for (int c = 0; c < CC; ++c) {
      float hv = hp[c];
#pragma unroll
      for (int j = 0; j < 16; ++j) t[j] += hv * read_W1[c * 16 + j];
    }
#pragma unroll
    for (int j = 0; j < 16; ++j) ea += silu_f(t[j]) * read_W2[j];
  }
  e_node[n] += ea;
}

__global__ void k_energy(const float* __restrict__ e_node, const int* __restrict__ batch,
                         float* __restrict__ energy) {
  int n = blockIdx.x * blockDim.x + threadIdx.x;
  if (n >= NN) return;
  atomicAdd(&energy[batch[n]], e_node[n]);
}

// ---------------------------------------------------------------------------
extern "C" void kernel_launch(void* const* d_in, const int* in_sizes, int n_in,
                              void* d_out, int out_size, void* d_ws, size_t ws_size,
                              hipStream_t stream) {
  const float* positions       = (const float*)d_in[0];
  const float* node_attrs      = (const float*)d_in[1];
  const float* shifts          = (const float*)d_in[2];
  const float* atomic_energies = (const float*)d_in[3];
  const float* W_embed         = (const float*)d_in[4];
  const float* Wr1             = (const float*)d_in[5];
  const float* Wr2             = (const float*)d_in[6];
  const float* Wr3             = (const float*)d_in[7];
  const float* Wprod           = (const float*)d_in[8];
  const float* Wq              = (const float*)d_in[9];
  const float* w_read0         = (const float*)d_in[10];
  const float* read_W1         = (const float*)d_in[11];
  const float* read_W2         = (const float*)d_in[12];
  const int*   edge_index      = (const int*)d_in[13];
  const int*   batch           = (const int*)d_in[14];

  float* out    = (float*)d_out;
  float* energy = out;            // [4]
  float* e_node = out + 4;        // [N]
  float* q      = out + 4 + NN;   // [N]

  char* ws = (char*)d_ws;
  float* m = (float*)ws;                                   // N*576 f32 (11.52 MB, fits L2)
  float* h = (float*)(ws + (size_t)NN * 576 * 4);          // N*64 f32
  _Float16* Wr1h   = (_Float16*)(ws + 12800000);           // 32B-aligned f16 weight pool
  _Float16* Wr2h   = Wr1h + 2 * 8 * 64;
  _Float16* Wr3h   = Wr2h + 2 * 64 * 64;
  _Float16* Wprodh = Wr3h + 2 * 64 * 192;

  k_prep<<<96, 256, 0, stream>>>(Wr1, Wr2, Wr3, Wprod, Wr1h, Wr2h, Wr3h, Wprodh);
  k_embed<<<(NN * CC + 255) / 256, 256, 0, stream>>>(node_attrs, W_embed, atomic_energies,
                                                     h, energy, e_node, q);
  for (int it = 0; it < 2; ++it) {
    k_zero4<<<(NN * 576 / 4 + 255) / 256, 256, 0, stream>>>((float4*)m, NN * 576 / 4);
    k_edge<<<(EE + 127) / 128, 256, 0, stream>>>(positions, shifts, edge_index, h,
                                                 Wr1h + it * 8 * 64, Wr2h + it * 64 * 64,
                                                 Wr3h + it * 64 * 192, m);
    k_node<<<(NN + 127) / 128, 256, 0, stream>>>(m, Wprodh + it * 192 * 64, h, it);
    k_readout<<<(NN + 255) / 256, 256, 0, stream>>>(h, w_read0, read_W1, read_W2,
                                                    Wq + it * 64, e_node, q, it);
  }
  k_energy<<<(NN + 255) / 256, 256, 0, stream>>>(e_node, batch, energy);
}
